// CARAFE_3_3_se_sa_norm_206158430924
// MI455X (gfx1250) — matrix operand order repaired and checked by reading, not compiled, not run on hardware
//
#include <hip/hip_runtime.h>
#include <hip/hip_bf16.h>

typedef __bf16 bf16;
typedef __attribute__((ext_vector_type(16))) __bf16 v16bf;
typedef __attribute__((ext_vector_type(8)))  float  v8f;

#define HW   16384              // 128*128
#define NBATCH 4
#define NPIX (NBATCH * HW)      // 65536 pixels total
#define NTILES (NPIX / 16)      // 4096 WMMA N-tiles

// ---------------------------------------------------------------------------
// A-matrix 16-bit fragment element -> K index (order chosen by this code, matches B loads; 16x32 A)
// lane 0-15: element e = K e (K0..15) ; lane 16-31: K16..31 (contiguous)
// element e of v16bf lives in VGPR e/2, low/high half by e&1.
// ---------------------------------------------------------------------------
__device__ __forceinline__ int a_klocal(int lane, int e) {
  int vg = e >> 1;
  (void)vg;
  return ((lane >> 4) << 4) + e;
}

// ---------------------------------------------------------------------------
// BN fold: scale = gamma * rsqrt(var+eps), bias = beta - mean*scale.
// Padded channels get scale=bias=0 so padded outputs are exactly zero.
// ---------------------------------------------------------------------------
__global__ void bn_prep(const float* __restrict__ bn1,   // [4,64]
                        const float* __restrict__ bn2,   // [4,100]
                        const float* __restrict__ bn3,   // [4,100]
                        float* __restrict__ s1, float* __restrict__ c1,
                        float* __restrict__ s2, float* __restrict__ c2,
                        float* __restrict__ s3, float* __restrict__ c3) {
  const float eps = 1e-5f;
  for (int i = threadIdx.x; i < 64; i += blockDim.x) {
    float g = bn1[i], be = bn1[64 + i], m = bn1[128 + i], v = bn1[192 + i];
    float s = g * rsqrtf(v + eps); s1[i] = s; c1[i] = be - m * s;
  }
  for (int i = threadIdx.x; i < 128; i += blockDim.x) {
    if (i < 100) {
      float g = bn2[i], be = bn2[100 + i], m = bn2[200 + i], v = bn2[300 + i];
      float s = g * rsqrtf(v + eps); s2[i] = s; c2[i] = be - m * s;
    } else { s2[i] = 0.f; c2[i] = 0.f; }
  }
  for (int i = threadIdx.x; i < 112; i += blockDim.x) {
    if (i < 100) {
      float g = bn3[i], be = bn3[100 + i], m = bn3[200 + i], v = bn3[300 + i];
      float s = g * rsqrtf(v + eps); s3[i] = s; c3[i] = be - m * s;
    } else { s3[i] = 0.f; c3[i] = 0.f; }
  }
}

// ---------------------------------------------------------------------------
// X [b,c,hw] f32  ->  Xt [b,hw,c] bf16   (pixel-major so B-fragments are
// contiguous 32B per lane in the conv1 GEMM). Reads fully coalesced.
// ---------------------------------------------------------------------------
__global__ __launch_bounds__(256) void x_to_bf16(const float* __restrict__ X,
                                                 bf16* __restrict__ Xt) {
  size_t t = (size_t)blockIdx.x * 256 + threadIdx.x;   // 16,777,216 total
  int b = (int)(t >> 22);
  int c = (int)(t >> 14) & 255;
  int p = (int)(t & 16383);
  Xt[((size_t)((b << 14) + p)) * 256 + c] = (bf16)X[t];
}

// ---------------------------------------------------------------------------
// Weight packing into WMMA A-fragment order: frag(mt,ks) -> 32 lanes x 16 bf16
// ---------------------------------------------------------------------------
__global__ void pack_comp(const float* __restrict__ w, bf16* __restrict__ ap) {
  int idx = blockIdx.x * 256 + threadIdx.x;            // 4*8*512 = 16384
  int e = idx & 15, lane = (idx >> 4) & 31, frag = idx >> 9;
  int mt = frag >> 3, ks = frag & 7;
  int m = mt * 16 + (lane & 15);
  int k = ks * 32 + a_klocal(lane, e);
  ap[idx] = (bf16)w[m * 256 + k];
}

__global__ void pack_enc(const float* __restrict__ w, bf16* __restrict__ ap) {
  int idx = blockIdx.x * 256 + threadIdx.x;            // 7*18*512 = 64512
  if (idx >= 7 * 18 * 512) return;
  int e = idx & 15, lane = (idx >> 4) & 31, frag = idx >> 9;
  int mt = frag / 18, ks = frag % 18;
  int m = mt * 16 + (lane & 15);
  int k = ks * 32 + a_klocal(lane, e);
  int tap = k >> 6, c = k & 63;                        // K = tap*64 + c
  float v = (m < 100) ? w[(m * 64 + c) * 9 + tap] : 0.f;
  ap[idx] = (bf16)v;
}

__global__ void pack_enc2(const float* __restrict__ w, bf16* __restrict__ ap) {
  int idx = blockIdx.x * 256 + threadIdx.x;            // 7*36*512 = 129024
  if (idx >= 7 * 36 * 512) return;
  int e = idx & 15, lane = (idx >> 4) & 31, frag = idx >> 9;
  int mt = frag / 36, ks = frag % 36;
  int m = mt * 16 + (lane & 15);
  int k = ks * 32 + a_klocal(lane, e);
  int tap = k >> 7, c = k & 127;                       // K = tap*128 + c (pad)
  float v = (m < 100 && c < 100) ? w[(m * 100 + c) * 9 + tap] : 0.f;
  ap[idx] = (bf16)v;
}

// ---------------------------------------------------------------------------
// WMMA conv GEMM.  One wave: MT 16-row tiles x NT 16-pixel N tiles (N=16*NT).
// Each A fragment feeds NT WMMAs -> halved A traffic, independent work for
// the scheduler between load and consumers.
// TAPS3: 3x3 conv, k-step -> (tap, 32-ch slice); tap shifts pixel pointer.
// Epilogue: BN scale/bias (+ReLU), store bf16 pixel-major [pix, COUT].
// ---------------------------------------------------------------------------
template<int MT, int NT, int KSTEPS, int CPT, int CIN, int COUT, bool TAPS3, bool RELU>
__global__ __launch_bounds__(256) void conv_gemm(
    const bf16* __restrict__ apack, const bf16* __restrict__ actin,
    bf16* __restrict__ actout,
    const float* __restrict__ scale, const float* __restrict__ bias) {
  int lane  = threadIdx.x & 31;
  int wv    = threadIdx.x >> 5;
  int nbase = (blockIdx.x * 8 + wv) * NT;          // first N-tile of this wave
  int khalf = (lane >> 4) << 4;                    // B fragment k-offset 0/16

  int gp[NT], bb[NT], py[NT], px[NT];
  #pragma unroll
  for (int t = 0; t < NT; ++t) {
    gp[t] = (nbase + t) * 16 + (lane & 15);
    bb[t] = gp[t] >> 14;
    int p = gp[t] & 16383;
    py[t] = p >> 7; px[t] = p & 127;
  }

  v8f acc[MT][NT];
  #pragma unroll
  for (int mt = 0; mt < MT; ++mt)
    #pragma unroll
    for (int t = 0; t < NT; ++t)
      #pragma unroll
      for (int r = 0; r < 8; ++r) acc[mt][t][r] = 0.f;

  for (int ks = 0; ks < KSTEPS; ++ks) {
    int tapdy = 0, tapdx = 0, cbase;
    if (TAPS3) {
      int tap = ks / CPT;
      cbase = (ks % CPT) * 32;
      tapdy = tap / 3 - 1; tapdx = tap % 3 - 1;
    } else cbase = ks * 32;

    v16bf bfrag[NT];
    #pragma unroll
    for (int t = 0; t < NT; ++t) {
      bool ok = true; int pp;
      if (TAPS3) {
        int ny = py[t] + tapdy, nx = px[t] + tapdx;
        ok = ((unsigned)ny < 128u) && ((unsigned)nx < 128u);
        pp = ((ny & 127) << 7) | (nx & 127);
      } else pp = (py[t] << 7) | px[t];
      #pragma unroll
      for (int i = 0; i < 16; ++i) bfrag[t][i] = (bf16)0.0f;
      if (ok)
        bfrag[t] = *(const v16bf*)(actin +
                     ((size_t)((bb[t] << 14) + pp) * CIN + cbase + khalf));
    }

    // pull next k-step's A fragments toward the cache (global_prefetch)
    if (ks + 1 < KSTEPS)
      __builtin_prefetch(apack + ((size_t)((ks + 1) * 32 + lane)) * 16, 0, 3);

    #pragma unroll
    for (int mt = 0; mt < MT; ++mt) {
      v16bf afrag = *(const v16bf*)(apack +
                      ((size_t)((mt * KSTEPS + ks) * 32 + lane)) * 16);
      #pragma unroll
      for (int t = 0; t < NT; ++t)
        acc[mt][t] = __builtin_amdgcn_wmma_f32_16x16x32_bf16(
            false, afrag, false, bfrag[t], (short)0, acc[mt][t], false, false);
    }
  }

  int mhalf = (lane >> 4) << 3;                    // D rows 0-7 / 8-15
  #pragma unroll
  for (int mt = 0; mt < MT; ++mt) {
    #pragma unroll
    for (int t = 0; t < NT; ++t) {
      #pragma unroll
      for (int r = 0; r < 8; ++r) {
        int m = mt * 16 + mhalf + r;
        float v = acc[mt][t][r] * scale[m] + bias[m];
        if (RELU) v = fmaxf(v, 0.f);
        actout[(size_t)gp[t] * COUT + m] = (bf16)v;
      }
    }
  }
}

// ---------------------------------------------------------------------------
// SE: per-(b,ch) spatial mean of enc2 output
// ---------------------------------------------------------------------------
__global__ __launch_bounds__(256) void se_reduce(const bf16* __restrict__ act3,
                                                 float* __restrict__ mean) {
  int b = blockIdx.x, ch = blockIdx.y;
  __shared__ float sh[256];
  float s = 0.f;
  for (int p = threadIdx.x; p < HW; p += 256)
    s += (float)act3[((size_t)(b * HW + p)) * 112 + ch];
  sh[threadIdx.x] = s; __syncthreads();
  for (int o = 128; o > 0; o >>= 1) {
    if (threadIdx.x < o) sh[threadIdx.x] += sh[threadIdx.x + o];
    __syncthreads();
  }
  if (threadIdx.x == 0) mean[b * 112 + ch] = sh[0] * (1.f / (float)HW);
}

__global__ void se_gate(const float* __restrict__ mean,
                        const float* __restrict__ w1, const float* __restrict__ b1,
                        const float* __restrict__ w2, const float* __restrict__ b2,
                        float* __restrict__ gate) {
  int b = blockIdx.x, t = threadIdx.x;
  __shared__ float h[6];
  if (t < 6) {
    float a = b1[t];
    for (int c = 0; c < 100; ++c) a += mean[b * 112 + c] * w1[t * 100 + c];
    h[t] = fmaxf(a, 0.f);
  }
  __syncthreads();
  if (t < 112) {
    float g = 0.f;
    if (t < 100) {
      float a = b2[t];
      for (int j = 0; j < 6; ++j) a += h[j] * w2[t * 6 + j];
      g = 1.f / (1.f + __expf(-a));
    }
    gate[b * 112 + t] = g;
  }
}

// ---------------------------------------------------------------------------
// Final CARAFE pass. Spatial attention + the 2x2 aliasing factors are common
// positive scalars across the 25 taps at a fine pixel -> cancel in the L1
// normalization, so only the per-channel SE gate is applied.
// One wave = 32 consecutive fine-x pixels; weights normalized once, reused
// across all 256 channels. Output stores fully coalesced.
// ---------------------------------------------------------------------------
__global__ __launch_bounds__(256) void carafe_out(const float* __restrict__ X,
                                                  const bf16* __restrict__ act3,
                                                  const float* __restrict__ gate,
                                                  float* __restrict__ out) {
  int lane = threadIdx.x & 31, wv = threadIdx.x >> 5;
  int y_ = blockIdx.x;                  // 0..255
  int b  = blockIdx.y;                  // 0..3
  int x_ = wv * 32 + lane;              // 0..255
  int cy = y_ >> 1, cx = x_ >> 1;
  int off = ((y_ & 1) << 1) | (x_ & 1);
  size_t pbase = ((size_t)(b * HW + cy * 128 + cx)) * 112;

  float w[25]; float s = 0.f;
  #pragma unroll
  for (int k = 0; k < 25; ++k) {
    int ch = k * 4 + off;
    float v = (float)act3[pbase + ch] * gate[b * 112 + ch];
    w[k] = v; s += fabsf(v);
  }
  float inv = 1.f / fmaxf(s, 1e-12f);

  int aoff[25]; float we[25];
  #pragma unroll
  for (int k = 0; k < 25; ++k) {
    int dy = k / 5 - 2, dx = k % 5 - 2;
    int sy = cy + dy, sx = cx + dx;
    bool okr = (unsigned)sy < 128u, okc = (unsigned)sx < 128u;
    int csy = min(max(sy, 0), 127), csx = min(max(sx, 0), 127);
    aoff[k] = (csy << 7) + csx;
    we[k]   = (okr && okc) ? w[k] * inv : 0.f;
  }

  const float* xb = X + (size_t)b * 256 * HW;
  float* ob = out + (size_t)b * 256 * 65536 + (size_t)y_ * 256 + x_;
  for (int c = 0; c < 256; ++c) {
    const float* xp = xb + (size_t)c * HW;
    float acc = 0.f;
    #pragma unroll
    for (int k = 0; k < 25; ++k) acc += we[k] * xp[aoff[k]];
    ob[(size_t)c * 65536] = acc;
  }
}

// ---------------------------------------------------------------------------
extern "C" void kernel_launch(void* const* d_in, const int* in_sizes, int n_in,
                              void* d_out, int out_size, void* d_ws, size_t ws_size,
                              hipStream_t stream) {
  const float* X       = (const float*)d_in[0];
  const float* comp_w  = (const float*)d_in[1];
  const float* comp_bn = (const float*)d_in[2];
  const float* enc_w   = (const float*)d_in[3];
  const float* enc_bn  = (const float*)d_in[4];
  const float* enc2_w  = (const float*)d_in[5];
  const float* enc2_bn = (const float*)d_in[6];
  const float* se_w1   = (const float*)d_in[7];
  const float* se_b1   = (const float*)d_in[8];
  const float* se_w2   = (const float*)d_in[9];
  const float* se_b2   = (const float*)d_in[10];
  float* out = (float*)d_out;

  // workspace layout (256B aligned regions)
  char* ws = (char*)d_ws;
  size_t o = 0;
  auto take = [&](size_t bytes) { size_t r = o; o += (bytes + 255) & ~(size_t)255; return r; };
  bf16*  Xt    = (bf16*)(ws + take((size_t)NPIX * 256 * 2));   // 33.6 MB
  bf16*  A1    = (bf16*)(ws + take(4 * 8 * 512 * 2));
  bf16*  A2    = (bf16*)(ws + take(7 * 18 * 512 * 2));
  bf16*  A3    = (bf16*)(ws + take(7 * 36 * 512 * 2));
  bf16*  act1  = (bf16*)(ws + take((size_t)NPIX * 64 * 2));    //  8.4 MB
  bf16*  act2  = (bf16*)(ws + take((size_t)NPIX * 128 * 2));   // 16.8 MB
  bf16*  act3  = (bf16*)(ws + take((size_t)NPIX * 112 * 2));   // 14.7 MB
  float* s1    = (float*)(ws + take(64 * 4));
  float* c1    = (float*)(ws + take(64 * 4));
  float* s2    = (float*)(ws + take(128 * 4));
  float* c2    = (float*)(ws + take(128 * 4));
  float* s3    = (float*)(ws + take(112 * 4));
  float* c3    = (float*)(ws + take(112 * 4));
  float* mean  = (float*)(ws + take(4 * 112 * 4));
  float* gate  = (float*)(ws + take(4 * 112 * 4));

  bn_prep<<<1, 128, 0, stream>>>(comp_bn, enc_bn, enc2_bn, s1, c1, s2, c2, s3, c3);
  x_to_bf16<<<65536, 256, 0, stream>>>(X, Xt);
  pack_comp<<<64, 256, 0, stream>>>(comp_w, A1);
  pack_enc <<<252, 256, 0, stream>>>(enc_w,  A2);
  pack_enc2<<<504, 256, 0, stream>>>(enc2_w, A3);

  // channels 112..127 of act2 are never written by the GEMM -> zero them all
  hipMemsetAsync(act2, 0, (size_t)NPIX * 128 * 2, stream);

  // 4096 N-tiles, NT=2 per wave, 8 waves per block -> 256 blocks per conv
  // conv1: 1x1 256->64, BN+ReLU       M=64  K=256  (8 ksteps)
  conv_gemm<4, 2, 8, 8, 256, 64, false, true><<<256, 256, 0, stream>>>(A1, Xt, act1, s1, c1);
  // enc:  3x3 64->100(pad112), BN     K=9*64=576 (18 ksteps), out stride 128
  conv_gemm<7, 2, 18, 2, 64, 128, true, false><<<256, 256, 0, stream>>>(A2, act1, act2, s2, c2);
  // enc2: 3x3 100->100(pad112), BN    K=9*128=1152 (36 ksteps, cin padded)
  conv_gemm<7, 2, 36, 4, 128, 112, true, false><<<256, 256, 0, stream>>>(A3, act2, act3, s3, c3);

  se_reduce<<<dim3(4, 100), 256, 0, stream>>>(act3, mean);
  se_gate<<<4, 128, 0, stream>>>(mean, se_w1, se_b1, se_w2, se_b2, gate);

  carafe_out<<<dim3(256, 4), 256, 0, stream>>>(X, act3, gate, out);

  (void)in_sizes; (void)n_in; (void)out_size; (void)ws_size;
}
